// SparseSpikingConv2D_68195490726057
// MI455X (gfx1250) — matrix operand order repaired and checked by reading, not compiled, not run on hardware
//
#include <hip/hip_runtime.h>

// SparseSpikingConv2D fused kernel for MI455X (gfx1250, wave32, WMMA).
// C[co][pixel] = W^T (f16, pre-swizzled) x im2col(x) (f16 from LDS), f32 accum,
// fused LIF membrane update + spike threshold epilogue.
// LDS x-tile stored ci-contiguous so B fragments load as two 16B v8h at ci+0 and ci+16 (ds_load_b128 pair).
// Staging uses float4 global loads (coalesced along w) + f16 scatter into the transposed tile.

typedef __attribute__((ext_vector_type(16))) _Float16 v16h;
typedef __attribute__((ext_vector_type(8)))  float    v8f;
typedef __attribute__((ext_vector_type(8)))  _Float16 v8h;

#define BS    16
#define CIN   64
#define COUT  128
#define HH    128
#define WW    128
#define KTOT  576          // CIN * 3 * 3
#define KSTEP 32
#define NSTEP 18           // KTOT / KSTEP
#define EPSF  1e-8f
#define XCOLS 130          // padded columns: c=0 and c=129 are zero borders

// A-matrix (16x32 f16) K index for (lane, half i) per CDNA5 ISA 7.12.2:
// VGPR j: lanes 0-15 hold K=2j,2j+1 ; lanes 16-31 hold K=8+2j,9+2j ; VGPR4-7 -> +16.
__device__ __forceinline__ int kmapA(int lane, int i) {
  int j  = i >> 1;
  int p  = i & 1;
  int g  = (j >> 2) << 4;          // 0 or 16
  int jj = j & 3;
  int hi = (lane & 16) ? 8 : 0;
  return g + hi + (jj << 1) + p;
}

// ---------------- prep: swizzle weights into A-fragment-ready f16 ----------------
__global__ void prep_weights(const float* __restrict__ kw, _Float16* __restrict__ wswz) {
  int idx = blockIdx.x * blockDim.x + threadIdx.x;   // [t][m16][lane][i]
  int i    = idx & 15;
  int lane = (idx >> 4) & 31;
  int m16  = (idx >> 9) & 7;
  int t    = idx >> 12;                              // 0..17
  int K    = t * KSTEP + kmapA(lane, i);             // flat (tap*64 + cin)
  int co   = m16 * 16 + (lane & 15);                 // A-matrix M = lane%16
  wswz[idx] = (_Float16)kw[K * COUT + co];
}

// ---------------- prep: per-cout kernel energy ----------------
__global__ void prep_norm(const float* __restrict__ kw, const float* __restrict__ bvec,
                          float* __restrict__ invn, float* __restrict__ bn) {
  int co = threadIdx.x;
  float s = 0.f;
  for (int K = 0; K < KTOT; ++K) {
    float v = kw[K * COUT + co];
    s += v * v;
  }
  invn[co] = 1.0f / (s + EPSF);
  bn[co]   = bvec[co] * s;
}

// ---------------- fused conv + LIF ----------------
__launch_bounds__(256)
__global__ void spiking_conv(const float* __restrict__ x,
                             const float* __restrict__ memin,
                             const _Float16* __restrict__ wswz,
                             const float* __restrict__ betap,
                             const float* __restrict__ bvec,
                             const float* __restrict__ invn,
                             const float* __restrict__ bn,
                             float* __restrict__ out) {
  // Transposed tile: xt[(dy*XCOLS + c)*CIN + ci]  (ci contiguous -> 32B B-frag loads)
  __shared__ _Float16 xt[3 * XCOLS * CIN];   // 49,920 B

  const int tid  = threadIdx.x;
  const int lane = tid & 31;                // wave32
  const int wid  = tid >> 5;                // 8 waves
  const int bimg = blockIdx.x >> 7;         // image
  const int h    = blockIdx.x & (HH - 1);   // output row

  // ---- zero borders (c=0 and c=129 for each dy,ci) ----
  for (int p = tid; p < 3 * CIN * 2; p += 256) {
    int dy   = p / (CIN * 2);
    int r    = p - dy * (CIN * 2);
    int side = r >> 6;                      // 0 -> c=0, 1 -> c=129
    int ci   = r & (CIN - 1);
    xt[(dy * XCOLS + (side ? 129 : 0)) * CIN + ci] = (_Float16)0.f;
  }

  // ---- stage x rows (h-1,h,h+1) as f16, transposed; float4 coalesced loads ----
  // q = (ci*3 + dy)*32 + v4 ; consecutive threads -> consecutive w within a row.
  for (int q = tid; q < CIN * 3 * (WW / 4); q += 256) {
    int row = q >> 5;                       // 0..191 = ci*3 + dy
    int v4  = q & 31;                       // float4 index along w
    int ci  = row / 3;
    int dy  = row - ci * 3;
    int hr  = h - 1 + dy;
    float4 val = make_float4(0.f, 0.f, 0.f, 0.f);
    if (hr >= 0 && hr < HH)
      val = *(const float4*)(x + ((size_t)(bimg * CIN + ci) * HH + hr) * WW + v4 * 4);
    const int base = (dy * XCOLS + 1 + v4 * 4) * CIN + ci;   // c = 1 + w
    xt[base          ] = (_Float16)val.x;
    xt[base + CIN    ] = (_Float16)val.y;
    xt[base + 2 * CIN] = (_Float16)val.z;
    xt[base + 3 * CIN] = (_Float16)val.w;
  }
  __syncthreads();

  const int wave_m = wid >> 2;              // 0..1 : 64 couts each
  const int wave_n = wid & 3;               // 0..3 : 32 pixels each
  const int nbase  = wave_n * 32;
  const int pcol   = nbase + (lane & 15);   // B-matrix N = lane%16 -> pixel w
  const int lhi16  = (lane & 16) >> 1;      // B: lanes 16-31 hold K+8 (and K+24)

  v8f acc[4][2] = {};

  #pragma unroll 1
  for (int t = 0; t < NSTEP; ++t) {
    const int tap    = t >> 1;              // 0..8
    const int dy     = tap / 3;
    const int dx     = tap - dy * 3;
    const int cibase = (t & 1) * 32 + lhi16;

    // A fragments: one aligned 32B global load each (L2-resident, pre-swizzled)
    v16h a[4];
    #pragma unroll
    for (int mi = 0; mi < 4; ++mi) {
      int m16 = wave_m * 4 + mi;
      a[mi] = *(const v16h*)(wswz + ((size_t)(t * 8 + m16) * 32 + lane) * 16);
    }
    if (t + 1 < NSTEP)
      __builtin_prefetch((const void*)(wswz + ((size_t)((t + 1) * 8 + wave_m * 4) * 32 + lane) * 16), 0, 1);

    // B fragments: two 16B-aligned v8h LDS loads each (ci-contiguous layout)
    v16h bfr[2];
    #pragma unroll
    for (int ni = 0; ni < 2; ++ni) {
      const int c = pcol + ni * 16 + dx;    // 0..129
      const _Float16* bp = xt + (size_t)(dy * XCOLS + c) * CIN + cibase;
      v8h blo = *(const v8h*)(bp);
      v8h bhi = *(const v8h*)(bp + 16);
      bfr[ni] = __builtin_shufflevector(blo, bhi, 0,1,2,3,4,5,6,7,8,9,10,11,12,13,14,15);
    }

    #pragma unroll
    for (int mi = 0; mi < 4; ++mi)
      #pragma unroll
      for (int ni = 0; ni < 2; ++ni)
        acc[mi][ni] = __builtin_amdgcn_wmma_f32_16x16x32_f16(
            false, a[mi], false, bfr[ni], (short)0, acc[mi][ni], false, false);
  }

  // ---- fused LIF epilogue: mem update, spike, reset ----
  const float beta = betap[0];
  const float omb  = 1.0f - beta;
  const size_t TOT = (size_t)BS * COUT * HH * WW;

  #pragma unroll
  for (int mi = 0; mi < 4; ++mi) {
    #pragma unroll
    for (int ni = 0; ni < 2; ++ni) {
      const int w = nbase + ni * 16 + (lane & 15);   // consecutive lanes -> consecutive w
      #pragma unroll
      for (int r = 0; r < 8; ++r) {
        const int co = wave_m * 64 + mi * 16 + ((lane >> 4) << 3) + r;
        const size_t idx = (((size_t)bimg * COUT + co) * HH + h) * WW + w;
        const float conv = acc[mi][ni][r];
        const float nm   = memin[idx] * beta + conv * omb;
        const float mthr = nm * invn[co] - bvec[co];
        const float spk  = mthr > 0.f ? 1.f : 0.f;
        out[idx]       = spk;
        out[TOT + idx] = nm - spk * bn[co];
      }
    }
  }
}

extern "C" void kernel_launch(void* const* d_in, const int* in_sizes, int n_in,
                              void* d_out, int out_size, void* d_ws, size_t ws_size,
                              hipStream_t stream) {
  const float* x    = (const float*)d_in[0];   // [16,64,128,128]
  const float* mem  = (const float*)d_in[1];   // [16,128,128,128]
  const float* kw   = (const float*)d_in[2];   // [576,128]
  const float* beta = (const float*)d_in[3];   // [1]
  const float* bvec = (const float*)d_in[4];   // [128]
  float* out = (float*)d_out;

  _Float16* wswz = (_Float16*)d_ws;                          // 73728 halves = 147456 B
  float* invn = (float*)((char*)d_ws + NSTEP * 8 * 32 * 16 * 2);
  float* bn   = invn + COUT;

  prep_weights<<<NSTEP * 8 * 32 * 16 / 256, 256, 0, stream>>>(kw, wswz);
  prep_norm<<<1, COUT, 0, stream>>>(kw, bvec, invn, bn);
  spiking_conv<<<BS * HH, 256, 0, stream>>>(x, mem, wswz, beta, bvec, invn, bn, out);
}
